// ABMIL_MB_41334765256856
// MI455X (gfx1250) — compile-verified, matrix operand order checked by reading, not run on hardware
//
#include <hip/hip_runtime.h>
#include <hip/hip_bf16.h>

// ---------------------------------------------------------------------------
// ABMIL multi-branch forward for MI455X (gfx1250), bf16 WMMA + f32 accumulate
// 80 rows/block; weights pre-swizzled into WMMA-fragment order; A-fragments
// batch-loaded per k-step so WMMAs issue back-to-back off one LDS wait.
// ---------------------------------------------------------------------------

#define NROWS 100000
#define DIMV  512
#define DV    256
#define RT    5          // row-tiles (of 16 rows) per block -> 80 rows
#define ROWSB (RT * 16)  // 80; 100000 = 1250 * 80

typedef __attribute__((ext_vector_type(16))) __bf16 v16bf;
typedef __attribute__((ext_vector_type(8)))  float  v8f;

union BF16Frag {
    v16bf          v;
    unsigned short u16[16];
    uint4          q[2];
};

__device__ __forceinline__ unsigned short f2bf(float f) {
    unsigned int u = __float_as_uint(f);
    u += 0x7FFFu + ((u >> 16) & 1u);     // round-to-nearest-even to bf16
    return (unsigned short)(u >> 16);
}

// A fragment from row-major bf16 LDS (ISA 7.12.2 16-bit layout):
// lane&15 = M row, lane>>4 = K half; elems 0..7 = k0+kh*8.., 8..15 = +16.
__device__ __forceinline__ v16bf frag_from_bf16(const unsigned short* rowk0, int lane) {
    const unsigned short* p = rowk0 + ((lane >> 4) & 1) * 8;
    BF16Frag f;
    f.q[0] = *(const uint4*)(p);         // 8 bf16 (ds_load_b128)
    f.q[1] = *(const uint4*)(p + 16);    // 8 bf16 (ds_load_b128)
    return f.v;
}

// B fragment from pre-swizzled weights: fragment = contiguous 1KB,
// lane ł owns 32 contiguous bytes -> two coalesced global_load_b128.
__device__ __forceinline__ v16bf frag_sw(const unsigned short* base, int fragIdx, int lane) {
    const unsigned short* p = base + ((size_t)fragIdx << 9) + (lane << 4);
    BF16Frag f;
    f.q[0] = *(const uint4*)(p);
    f.q[1] = *(const uint4*)(p + 8);
    return f.v;
}

__device__ __forceinline__ v8f wmma_bf16(v16bf a, v16bf b, v8f c) {
    return __builtin_amdgcn_wmma_f32_16x16x32_bf16(
        /*neg_a=*/false, a, /*neg_b=*/false, b,
        /*c_mod=*/(short)0, c, /*reuse_a=*/false, /*reuse_b=*/false);
}

// ---------------------------------------------------------------------------
// fp32 weights -> bf16 in WMMA-fragment-swizzled order.
// ---------------------------------------------------------------------------
__global__ __launch_bounds__(256) void swizzle_bf16_kernel(
    const float* __restrict__ src, unsigned short* __restrict__ dst, int NN, int K) {
    int o4 = blockIdx.x * blockDim.x + threadIdx.x;
    int total = NN * K / 8;
    if (o4 >= total) return;
    int half = o4 & 1;                 // which 8-slot half of the lane
    int lane = (o4 >> 1) & 31;
    int frag = o4 >> 6;
    int kblocks = K >> 5;
    int nt = frag / kblocks;
    int kb = frag - nt * kblocks;
    int row = nt * 16 + (lane & 15);
    int kh = (lane >> 4) & 1;
    int k = kb * 32 + kh * 8 + half * 16;
    const float* s = src + (size_t)row * K + k;
    float4 a = *(const float4*)(s);
    float4 b = *(const float4*)(s + 4);
    uint4 o;
    o.x = (unsigned)f2bf(a.x) | ((unsigned)f2bf(a.y) << 16);
    o.y = (unsigned)f2bf(a.z) | ((unsigned)f2bf(a.w) << 16);
    o.z = (unsigned)f2bf(b.x) | ((unsigned)f2bf(b.y) << 16);
    o.w = (unsigned)f2bf(b.z) | ((unsigned)f2bf(b.w) << 16);
    ((uint4*)dst)[o4] = o;
}

__global__ __launch_bounds__(256) void zero_kernel(float* __restrict__ p, int n) {
    for (int i = threadIdx.x; i < n; i += 256) p[i] = 0.f;
}

// ---------------------------------------------------------------------------
// Main kernel: 1 block = 80 rows (5 row-tiles), 8 waves, all 4 GEMMs fused.
// ---------------------------------------------------------------------------
__global__ __launch_bounds__(256) void abmil_rows_kernel(
    const float* __restrict__ h,
    const float* __restrict__ b1, const float* __restrict__ b2,
    const float* __restrict__ ba, const float* __restrict__ bbv,
    const float* __restrict__ Wc, const float* __restrict__ bc,
    const unsigned short* __restrict__ W1sw, const unsigned short* __restrict__ W2sw,
    const unsigned short* __restrict__ Wasw, const unsigned short* __restrict__ Wbsw,
    float* __restrict__ hnew_out, float* __restrict__ logits_out) {

    // hbf_s holds bf16 h tiles for GEMM1, then is overwritten (after a
    // barrier) with bf16 h' for GEMM3/4.
    __shared__ unsigned short hbf_s[RT][16][DIMV];   // 80 KB
    __shared__ unsigned short hid_s[RT][16][DV];     // 40 KB
    __shared__ float          logit_s[2][RT][16];    // 640 B

    const int tid   = threadIdx.x;
    const int lane  = tid & 31;
    const int wave  = tid >> 5;          // 0..7
    const int laneM = lane & 15;
    const int hh    = (lane >> 4) & 1;
    const int rowBase = blockIdx.x * ROWSB;
    const float* htile = h + (size_t)rowBase * DIMV;

    // ---- prologue: stage 80 rows of h as bf16 in LDS ----
    for (int i = tid; i < ROWSB * DIMV / 4; i += 256) {
        float4 v = ((const float4*)htile)[i];
        unsigned int lo = (unsigned int)f2bf(v.x) | ((unsigned int)f2bf(v.y) << 16);
        unsigned int hi = (unsigned int)f2bf(v.z) | ((unsigned int)f2bf(v.w) << 16);
        ((uint2*)&hbf_s[0][0][0])[i] = make_uint2(lo, hi);
    }
    if (tid < 2 * RT * 16) ((float*)logit_s)[tid] = 0.f;
    __syncthreads();

    // ---- GEMM1: hid = relu(h @ W1^T + b1), 80x256, K=512 ----
    // Both column-tiles accumulated in one k-pass so each af[] batch feeds
    // 10 WMMAs (acc: 10 x v8f = 80 VGPRs).
    {
        v8f acc[2][RT] = {};
        for (int kb = 0; kb < DIMV / 32; ++kb) {
            v16bf af[RT];
#pragma unroll
            for (int t = 0; t < RT; ++t)
                af[t] = frag_from_bf16(&hbf_s[t][laneM][kb * 32], lane);
#pragma unroll
            for (int nt = 0; nt < 2; ++nt) {
                v16bf bf = frag_sw(W1sw, (wave * 2 + nt) * (DIMV / 32) + kb, lane);
#pragma unroll
                for (int t = 0; t < RT; ++t)
                    acc[nt][t] = wmma_bf16(af[t], bf, acc[nt][t]);
            }
        }
#pragma unroll
        for (int nt = 0; nt < 2; ++nt) {
            const int ncol = (wave * 2 + nt) * 16 + laneM;
            float bia = b1[ncol];
#pragma unroll
            for (int t = 0; t < RT; ++t)
#pragma unroll
                for (int r = 0; r < 8; ++r)
                    hid_s[t][r + 8 * hh][ncol] = f2bf(fmaxf(acc[nt][t][r] + bia, 0.f));
        }
    }
    __syncthreads();

    // ---- GEMM2: h' = h + hid @ W2^T + b2, 80x512, K=256 ----
    // 4 column-tiles per wave handled as two passes of 2-tile pairs.
#pragma unroll
    for (int ip = 0; ip < 2; ++ip) {
        v8f acc[2][RT] = {};
        for (int kb = 0; kb < DV / 32; ++kb) {
            v16bf af[RT];
#pragma unroll
            for (int t = 0; t < RT; ++t)
                af[t] = frag_from_bf16(&hid_s[t][laneM][kb * 32], lane);
#pragma unroll
            for (int i2 = 0; i2 < 2; ++i2) {
                const int ntg = wave * 4 + ip * 2 + i2;   // 0..31
                v16bf bf = frag_sw(W2sw, ntg * (DV / 32) + kb, lane);
#pragma unroll
                for (int t = 0; t < RT; ++t)
                    acc[i2][t] = wmma_bf16(af[t], bf, acc[i2][t]);
            }
        }
#pragma unroll
        for (int i2 = 0; i2 < 2; ++i2) {
            const int nc = (wave * 4 + ip * 2 + i2) * 16 + laneM;
            float bia = b2[nc];
#pragma unroll
            for (int t = 0; t < RT; ++t) {
#pragma unroll
                for (int r = 0; r < 8; ++r) {
                    int m = r + 8 * hh;
                    size_t row = (size_t)(rowBase + t * 16 + m);
                    float v = acc[i2][t][r] + bia + h[row * DIMV + nc]; // residual
                    hnew_out[row * DIMV + nc] = v;                      // fp32 h'
                    hbf_s[t][m][nc] = f2bf(v);                          // bf16 h'
                }
            }
        }
    }
    __syncthreads();

    // ---- GEMM3/4 per class: a = tanh(h'Wa^T+ba), g = sigmoid(h'Wb^T+bb),
    //      logit_c[row] += sum_k a*g*Wc[c,k] ----
    for (int c = 0; c < 2; ++c) {
        const unsigned short* WaC = Wasw + (size_t)c * DV * DIMV;
        const unsigned short* WbC = Wbsw + (size_t)c * DV * DIMV;
#pragma unroll
        for (int nt = 0; nt < 2; ++nt) {
            const int ntg  = wave * 2 + nt;
            const int ncol = ntg * 16 + laneM;
            v8f aA[RT] = {}, aG[RT] = {};
            for (int kb = 0; kb < DIMV / 32; ++kb) {
                v16bf af[RT];
#pragma unroll
                for (int t = 0; t < RT; ++t)
                    af[t] = frag_from_bf16(&hbf_s[t][laneM][kb * 32], lane);
                v16bf bA = frag_sw(WaC, ntg * (DIMV / 32) + kb, lane);
                v16bf bG = frag_sw(WbC, ntg * (DIMV / 32) + kb, lane);
#pragma unroll
                for (int t = 0; t < RT; ++t)
                    aA[t] = wmma_bf16(af[t], bA, aA[t]);
#pragma unroll
                for (int t = 0; t < RT; ++t)
                    aG[t] = wmma_bf16(af[t], bG, aG[t]);
            }
            float bAv = ba[c * DV + ncol];
            float bGv = bbv[c * DV + ncol];
            float wcv = Wc[c * DV + ncol];
#pragma unroll
            for (int t = 0; t < RT; ++t)
#pragma unroll
                for (int r = 0; r < 8; ++r) {
                    int m = r + 8 * hh;
                    float av = tanhf(aA[t][r] + bAv);
                    float gv = 1.f / (1.f + __expf(-(aG[t][r] + bGv)));
                    atomicAdd(&logit_s[c][t][m], av * gv * wcv);
                }
        }
    }
    __syncthreads();

    for (int i = tid; i < 2 * RT * 16; i += 256) {
        int c = i / (RT * 16), rem = i % (RT * 16), t = rem / 16, m = rem % 16;
        logits_out[(size_t)c * NROWS + rowBase + t * 16 + m] =
            logit_s[c][t][m] + bc[c];
    }
}

// ---------------------------------------------------------------------------
// Per-class softmax over N (one block per class); writes normalized A
// ---------------------------------------------------------------------------
__global__ __launch_bounds__(512) void softmax_kernel(
    const float* __restrict__ logits_ws, float* __restrict__ A_out, int n) {
    __shared__ float red[512];
    const int c = blockIdx.x;
    const int tid = threadIdx.x;
    const float* x = logits_ws + (size_t)c * n;
    float* y = A_out + (size_t)c * n;

    float mx = -3.0e38f;
    for (int i = tid; i < n; i += 512) mx = fmaxf(mx, x[i]);
    red[tid] = mx; __syncthreads();
    for (int s = 256; s > 0; s >>= 1) {
        if (tid < s) red[tid] = fmaxf(red[tid], red[tid + s]);
        __syncthreads();
    }
    mx = red[0]; __syncthreads();

    float sum = 0.f;
    for (int i = tid; i < n; i += 512) {
        float e = __expf(x[i] - mx);
        y[i] = e;
        sum += e;
    }
    red[tid] = sum; __syncthreads();
    for (int s = 256; s > 0; s >>= 1) {
        if (tid < s) red[tid] += red[tid + s];
        __syncthreads();
    }
    float inv = 1.f / red[0];
    for (int i = tid; i < n; i += 512) y[i] *= inv;
}

// ---------------------------------------------------------------------------
// M[c,:] = sum_n A[c,n] * h'[n,:]   (per-block partials + global atomics)
// ---------------------------------------------------------------------------
__global__ __launch_bounds__(256) void weighted_sum_kernel(
    const float* __restrict__ A, const float* __restrict__ hnew,
    float* __restrict__ Mout, int n) {
    const int tid = threadIdx.x;
    const int rows_per_blk = (n + gridDim.x - 1) / gridDim.x;
    const int r0 = blockIdx.x * rows_per_blk;
    const int r1 = min(n, r0 + rows_per_blk);
    const int c0 = tid * 2;                 // two columns per thread (512/256)

    float a00 = 0.f, a01 = 0.f, a10 = 0.f, a11 = 0.f;
    for (int r = r0; r < r1; ++r) {
        float w0 = A[r];
        float w1 = A[(size_t)n + r];
        float2 hv = *(const float2*)(hnew + (size_t)r * DIMV + c0);
        a00 += w0 * hv.x; a01 += w0 * hv.y;
        a10 += w1 * hv.x; a11 += w1 * hv.y;
    }
    atomicAdd(&Mout[c0],            a00);
    atomicAdd(&Mout[c0 + 1],        a01);
    atomicAdd(&Mout[DIMV + c0],     a10);
    atomicAdd(&Mout[DIMV + c0 + 1], a11);
}

// ---------------------------------------------------------------------------
// Final: logits, softmax over C=2, argmax, features = M[-1]
// ---------------------------------------------------------------------------
__global__ __launch_bounds__(512) void final_kernel(
    const float* __restrict__ Mws, const float* __restrict__ Wclf,
    const float* __restrict__ bclf, float* __restrict__ out) {
    __shared__ float red0[512];
    __shared__ float red1[512];
    const int tid = threadIdx.x;
    red0[tid] = Mws[tid]        * Wclf[tid];
    red1[tid] = Mws[DIMV + tid] * Wclf[DIMV + tid];
    __syncthreads();
    for (int s = 256; s > 0; s >>= 1) {
        if (tid < s) { red0[tid] += red0[tid + s]; red1[tid] += red1[tid + s]; }
        __syncthreads();
    }
    if (tid == 0) {
        float l0 = red0[0] + bclf[0];
        float l1 = red1[0] + bclf[1];
        float mx = fmaxf(l0, l1);
        float e0 = __expf(l0 - mx), e1 = __expf(l1 - mx);
        float inv = 1.f / (e0 + e1);
        out[0] = l0;       out[1] = l1;        // logits (1,2)
        out[2] = e0 * inv; out[3] = e1 * inv;  // Y_prob (1,2)
        out[4] = (l1 > l0) ? 1.0f : 0.0f;      // Y_hat  (1,1)
    }
    out[5 + 2 * NROWS + tid] = Mws[DIMV + tid]; // features = M[-1:] (1,512)
}

// ---------------------------------------------------------------------------
extern "C" void kernel_launch(void* const* d_in, const int* in_sizes, int n_in,
                              void* d_out, int out_size, void* d_ws, size_t ws_size,
                              hipStream_t stream) {
    (void)in_sizes; (void)n_in; (void)out_size; (void)ws_size;
    const float* h    = (const float*)d_in[0];
    const float* W1   = (const float*)d_in[1];
    const float* b1   = (const float*)d_in[2];
    const float* W2   = (const float*)d_in[3];
    const float* b2   = (const float*)d_in[4];
    const float* Wa   = (const float*)d_in[5];
    const float* ba   = (const float*)d_in[6];
    const float* Wb   = (const float*)d_in[7];
    const float* bbv  = (const float*)d_in[8];
    const float* Wc   = (const float*)d_in[9];
    const float* bc   = (const float*)d_in[10];
    const float* Wclf = (const float*)d_in[11];
    const float* bclf = (const float*)d_in[12];

    float* out = (float*)d_out;
    float* ws  = (float*)d_ws;

    // workspace layout (floats):
    //   [0, N*512)              h' (residual output, fp32)
    //   [N*512, N*512+2N)       per-class attention logits
    //   [.., +1024)             M accumulator (2 x 512)
    //   then swizzled bf16 weights: W1 | W2 | Wa | Wb
    float*          hnew_ws   = ws;
    float*          logits_ws = ws + (size_t)NROWS * DIMV;
    float*          M_ws      = logits_ws + 2 * (size_t)NROWS;
    unsigned short* W1sw      = (unsigned short*)(M_ws + 1024);
    unsigned short* W2sw      = W1sw + (size_t)DV * DIMV;
    unsigned short* Wasw      = W2sw + (size_t)DIMV * DV;
    unsigned short* Wbsw      = Wasw + (size_t)2 * DV * DIMV;

    // swizzle weights into fragment order (Wa/Wb: 2 classes flattened into NN)
    {
        int t1 = DV * DIMV / 8;        // W1: NN=256, K=512
        int t2 = DIMV * DV / 8;        // W2: NN=512, K=256
        int ta = 2 * DV * DIMV / 8;    // Wa/Wb: NN=512 (2*256), K=512
        swizzle_bf16_kernel<<<(t1 + 255) / 256, 256, 0, stream>>>(W1, W1sw, DV, DIMV);
        swizzle_bf16_kernel<<<(t2 + 255) / 256, 256, 0, stream>>>(W2, W2sw, DIMV, DV);
        swizzle_bf16_kernel<<<(ta + 255) / 256, 256, 0, stream>>>(Wa, Wasw, 2 * DV, DIMV);
        swizzle_bf16_kernel<<<(ta + 255) / 256, 256, 0, stream>>>(Wb, Wbsw, 2 * DV, DIMV);
    }
    zero_kernel<<<1, 256, 0, stream>>>(M_ws, 2 * DIMV);

    abmil_rows_kernel<<<NROWS / ROWSB, 256, 0, stream>>>(
        h, b1, b2, ba, bbv, Wc, bc, W1sw, W2sw, Wasw, Wbsw, hnew_ws, logits_ws);

    softmax_kernel<<<2, 512, 0, stream>>>(logits_ws, out + 5, NROWS);
    weighted_sum_kernel<<<256, 256, 0, stream>>>(out + 5, hnew_ws, M_ws, NROWS);
    final_kernel<<<1, 512, 0, stream>>>(M_ws, Wclf, bclf, out);
}
